// Mamba2Layer_23536420782158
// MI455X (gfx1250) — matrix operand order repaired and checked by reading, not compiled, not run on hardware
//
#include <hip/hip_runtime.h>
#include <hip/hip_bf16.h>
#include <math.h>

// ---------------------------------------------------------------------------
// Mamba2 layer forward for MI455X (gfx1250), wave32 + WMMA f16->f32.
// LDS tiles stored fragment-friendly (k innermost) so every WMMA operand is
// fetched with two ds_load_b128 per lane instead of 16 ds_load_u16.
// ---------------------------------------------------------------------------

typedef __attribute__((ext_vector_type(16))) _Float16 v16h;
typedef __attribute__((ext_vector_type(8)))  _Float16 v8h;
typedef __attribute__((ext_vector_type(4)))  _Float16 v4h;
typedef __attribute__((ext_vector_type(8)))  float    v8f;

#define BATCH   2
#define LSEQ    4096
#define DMODEL  1024
#define DSTATE  64
#define HEADDIM 128
#define CHUNK   256
#define DINNER  2048
#define NHEADS  16
#define CONVDIM 2176
#define DINPROJ 4240
#define NCHUNK  16
#define MTOT    (BATCH*LSEQ)      // 8192 rows for both big GEMMs

// -------- workspace layout (float elements) --------
#define OFF_PROJ  ((size_t)0)
#define SZ_PROJ   ((size_t)BATCH*LSEQ*DINPROJ)
#define OFF_DT    (OFF_PROJ + SZ_PROJ)
#define SZ_DT     ((size_t)BATCH*LSEQ*NHEADS)
#define OFF_ACS   (OFF_DT + SZ_DT)
#define SZ_ACS    ((size_t)BATCH*NCHUNK*NHEADS*CHUNK)
#define OFF_TOT   (OFF_ACS + SZ_ACS)
#define SZ_TOT    ((size_t)BATCH*NCHUNK*NHEADS)
#define OFF_ST    (OFF_TOT + SZ_TOT)
#define SZ_ST     ((size_t)BATCH*NCHUNK*NHEADS*HEADDIM*DSTATE)
#define OFF_Y     (OFF_ST + SZ_ST)

// ---------------------------------------------------------------------------
// WMMA helpers
// ---------------------------------------------------------------------------
static __device__ __forceinline__ v8f wmma_f16(v16h a, v16h b, v8f c) {
  return __builtin_amdgcn_wmma_f32_16x16x32_f16(false, a, false, b,
                                                (short)0, c, false, false);
}

static __device__ __forceinline__ v16h frag_cat(v8h lo, v8h hi) {
  return __builtin_shufflevector(lo, hi, 0, 1, 2, 3, 4, 5, 6, 7,
                                 8, 9, 10, 11, 12, 13, 14, 15);
}

// A fragment (16 rows x 32 k) from LDS tile stored [row][k], ld%8==0.
// lanes 0-15: K 0..7 + 16..23 ; lanes 16-31: K 8..15 + 24..31
static __device__ __forceinline__ v16h load_a_frag(const _Float16* base, int ld,
                                                   int row, int lane) {
  const int kh = (lane >> 4) * 8;
  const _Float16* p = base + row * ld;
  const v8h lo = *(const v8h*)(p + kh);
  const v8h hi = *(const v8h*)(p + 16 + kh);
  return frag_cat(lo, hi);
}

// B fragment (32 k x 16 cols) from LDS tile stored [col][k], ld%8==0.
// lanes 0-15: K 0..7 + 16..23 ; lanes 16-31: K 8..15 + 24..31
static __device__ __forceinline__ v16h load_b_frag(const _Float16* base, int ld,
                                                   int col, int lane) {
  const _Float16* p = base + col * ld + (lane >> 4) * 8;
  const v8h lo = *(const v8h*)(p);
  const v8h hi = *(const v8h*)(p + 16);
  return frag_cat(lo, hi);
}

static __device__ __forceinline__ v4h pack4(float4 v) {
  v4h r;
  r[0] = (_Float16)v.x; r[1] = (_Float16)v.y;
  r[2] = (_Float16)v.z; r[3] = (_Float16)v.w;
  return r;
}
static __device__ __forceinline__ v4h pack4s(float4 v, float s) {
  v4h r;
  r[0] = (_Float16)(v.x * s); r[1] = (_Float16)(v.y * s);
  r[2] = (_Float16)(v.z * s); r[3] = (_Float16)(v.w * s);
  return r;
}

// ---------------------------------------------------------------------------
// Kernel: C[M,N] = A[M,K] * W[N,K]^T   (fp32 in, f16 WMMA, fp32 out)
// Block tile 128x64, K step 32, 8 waves (4x2 layout of 32x32 sub-tiles).
// ---------------------------------------------------------------------------
__global__ __launch_bounds__(256)
void gemm_tn(const float* __restrict__ A, const float* __restrict__ W,
             float* __restrict__ C, int M, int N, int K) {
  __shared__ __align__(16) _Float16 As[128 * 40];  // [m][k] k-pad to 40
  __shared__ __align__(16) _Float16 Bs[64 * 40];   // [n][k]
  const int LD = 40;
  const int t = threadIdx.x, lane = t & 31, wave = t >> 5;
  const int bm = blockIdx.y * 128, bn = blockIdx.x * 64;
  const int wm = (wave >> 1) * 32, wn = (wave & 1) * 32;
  const int r0 = t >> 3, kc = (t & 7) * 4;

  v8f acc00 = {}, acc01 = {}, acc10 = {}, acc11 = {};

  for (int kt = 0; kt < K; kt += 32) {
    // stage A tile 128x32 (f32 -> f16), contiguous b64 LDS stores
#pragma unroll
    for (int i = 0; i < 4; ++i) {
      const int m = r0 + 32 * i;
      const float4 v = *(const float4*)(A + (size_t)(bm + m) * K + kt + kc);
      *(v4h*)(As + m * LD + kc) = pack4(v);
    }
    if (kt + 32 < K)  // hint next K slice (global_prefetch_b8)
      __builtin_prefetch(A + (size_t)(bm + r0) * K + kt + 32 + kc, 0, 0);
    // stage W tile 64x32
#pragma unroll
    for (int i = 0; i < 2; ++i) {
      const int n = r0 + 32 * i;
      const int ng = bn + n;
      float4 v = make_float4(0.f, 0.f, 0.f, 0.f);
      if (ng < N) v = *(const float4*)(W + (size_t)ng * K + kt + kc);
      *(v4h*)(Bs + n * LD + kc) = pack4(v);
    }
    __syncthreads();

    const v16h a0 = load_a_frag(As, LD, wm + (lane & 15), lane);
    const v16h a1 = load_a_frag(As, LD, wm + 16 + (lane & 15), lane);
    const v16h b0 = load_b_frag(Bs, LD, wn + (lane & 15), lane);
    const v16h b1 = load_b_frag(Bs, LD, wn + 16 + (lane & 15), lane);
    acc00 = wmma_f16(a0, b0, acc00);
    acc01 = wmma_f16(a0, b1, acc01);
    acc10 = wmma_f16(a1, b0, acc10);
    acc11 = wmma_f16(a1, b1, acc11);
    __syncthreads();
  }

  // C layout: lane 0-15 -> N=lane, M=r ; lane 16-31 -> N=lane-16, M=8+r
  const int col = bn + wn + (lane & 15);
  const int rbase = bm + wm + 8 * (lane >> 4);
#pragma unroll
  for (int r = 0; r < 8; ++r) {
    const int row = rbase + r;
    if (col < N) {
      C[(size_t)row * N + col]        = acc00[r];
      C[(size_t)(row + 16) * N + col] = acc10[r];
    }
    if (col + 16 < N) {
      C[(size_t)row * N + col + 16]        = acc01[r];
      C[(size_t)(row + 16) * N + col + 16] = acc11[r];
    }
  }
}

// ---------------------------------------------------------------------------
// depthwise causal conv (k=4) + bias + SiLU, in place on xBC slice.
// ---------------------------------------------------------------------------
__global__ __launch_bounds__(256)
void conv_silu_kernel(float* __restrict__ proj, const float* __restrict__ cw,
                      const float* __restrict__ cb) {
  const int id = blockIdx.x * blockDim.x + threadIdx.x;
  if (id >= BATCH * CONVDIM) return;
  const int b = id / CONVDIM, c = id % CONVDIM;
  const float w0 = cw[c * 4 + 0], w1 = cw[c * 4 + 1];
  const float w2 = cw[c * 4 + 2], w3 = cw[c * 4 + 3];
  const float bias = cb[c];
  float x0 = 0.f, x1 = 0.f, x2 = 0.f;
  size_t base = (size_t)b * LSEQ * DINPROJ + DINNER + c;
  for (int l = 0; l < LSEQ; ++l) {
    const float v = proj[base];
    float o = w0 * x0 + w1 * x1 + w2 * x2 + w3 * v + bias;
    o = o / (1.f + expf(-o));              // SiLU
    proj[base] = o;
    x0 = x1; x1 = x2; x2 = v;
    base += DINPROJ;
  }
}

// ---------------------------------------------------------------------------
// dt = softplus(dt_raw + dt_bias)
// ---------------------------------------------------------------------------
__global__ __launch_bounds__(256)
void dt_kernel(const float* __restrict__ proj, const float* __restrict__ dt_bias,
               float* __restrict__ dt) {
  const int id = blockIdx.x * blockDim.x + threadIdx.x;
  if (id >= BATCH * LSEQ * NHEADS) return;
  const int h = id & (NHEADS - 1);
  const int bl = id >> 4;  // b*L + l
  const float v = proj[(size_t)bl * DINPROJ + DINNER + CONVDIM + h] + dt_bias[h];
  dt[id] = (v > 20.f) ? v : log1pf(expf(v));
}

// ---------------------------------------------------------------------------
// per (b,chunk,head) cumsum of dt*A -> A_cs, chunk totals.
// ---------------------------------------------------------------------------
__global__ __launch_bounds__(256)
void cumsum_kernel(const float* __restrict__ dt, const float* __restrict__ A_log,
                   float* __restrict__ acs, float* __restrict__ total) {
  const int id = blockIdx.x * blockDim.x + threadIdx.x;
  if (id >= BATCH * NCHUNK * NHEADS) return;
  const int h = id & (NHEADS - 1);
  const int c = (id >> 4) & (NCHUNK - 1);
  const int b = id >> 8;
  const float A = -expf(A_log[h]);
  float acc = 0.f;
  const size_t dbase = ((size_t)b * LSEQ + (size_t)c * CHUNK) * NHEADS + h;
  const size_t abase = (size_t)id * CHUNK;
  for (int k = 0; k < CHUNK; ++k) {
    acc += dt[dbase + (size_t)k * NHEADS] * A;
    acs[abase + k] = acc;
  }
  total[id] = acc;
}

// ---------------------------------------------------------------------------
// chunk states: states[p,n] = sum_k x[k,p]*dt[k]*exp(tot-Acs[k])*B[k,n]
// One block per (b,chunk,head): WMMA 128x64 over K=256.
// ---------------------------------------------------------------------------
__global__ __launch_bounds__(256)
void states_kernel(const float* __restrict__ proj, const float* __restrict__ dt,
                   const float* __restrict__ acs, const float* __restrict__ total,
                   float* __restrict__ states) {
  __shared__ __align__(16) _Float16 xw[128 * 40];  // [p][k]
  __shared__ __align__(16) _Float16 Bs[64 * 40];   // [n][k]
  const int LD = 40;
  const int bch = blockIdx.x;               // (b*NCHUNK + c)*NHEADS + h
  const int h = bch & (NHEADS - 1);
  const int c = (bch >> 4) & (NCHUNK - 1);
  const int b = bch >> 8;
  const int t = threadIdx.x, lane = t & 31, wave = t >> 5;
  const float tot = total[bch];

  v8f acc[4] = {v8f{}, v8f{}, v8f{}, v8f{}};

  for (int kt = 0; kt < CHUNK; kt += 32) {
    const int kk = t >> 3;
    const int kabs = kt + kk;
    const size_t row = (size_t)(b * LSEQ + c * CHUNK + kabs) * DINPROJ;
    const float w = dt[((size_t)(b * LSEQ + c * CHUNK + kabs)) * NHEADS + h] *
                    expf(tot - acs[(size_t)bch * CHUNK + kabs]);
    // x * w  (16 p values per thread; scatter b16 stores into [p][k])
    {
      const int p0 = (t & 7) * 16;
#pragma unroll
      for (int i = 0; i < 4; ++i) {
        const float4 v = *(const float4*)(proj + row + DINNER + h * HEADDIM + p0 + 4 * i);
        xw[(p0 + 4 * i + 0) * LD + kk] = (_Float16)(v.x * w);
        xw[(p0 + 4 * i + 1) * LD + kk] = (_Float16)(v.y * w);
        xw[(p0 + 4 * i + 2) * LD + kk] = (_Float16)(v.z * w);
        xw[(p0 + 4 * i + 3) * LD + kk] = (_Float16)(v.w * w);
      }
    }
    // B matrix (8 n values per thread)
    {
      const int n0 = (t & 7) * 8;
#pragma unroll
      for (int i = 0; i < 2; ++i) {
        const float4 v = *(const float4*)(proj + row + 2 * DINNER + n0 + 4 * i);
        Bs[(n0 + 4 * i + 0) * LD + kk] = (_Float16)v.x;
        Bs[(n0 + 4 * i + 1) * LD + kk] = (_Float16)v.y;
        Bs[(n0 + 4 * i + 2) * LD + kk] = (_Float16)v.z;
        Bs[(n0 + 4 * i + 3) * LD + kk] = (_Float16)v.w;
      }
    }
    __syncthreads();
    const v16h a = load_a_frag(xw, LD, wave * 16 + (lane & 15), lane);
#pragma unroll
    for (int j = 0; j < 4; ++j) {
      const v16h bf = load_b_frag(Bs, LD, j * 16 + (lane & 15), lane);
      acc[j] = wmma_f16(a, bf, acc[j]);
    }
    __syncthreads();
  }

  const size_t sbase = (size_t)bch * HEADDIM * DSTATE;
#pragma unroll
  for (int j = 0; j < 4; ++j) {
    const int n = j * 16 + (lane & 15);
    const int pb = wave * 16 + 8 * (lane >> 4);
#pragma unroll
    for (int r = 0; r < 8; ++r)
      states[sbase + (size_t)(pb + r) * DSTATE + n] = acc[j][r];
  }
}

// ---------------------------------------------------------------------------
// inter-chunk recurrence, in place (states -> prev-states).
// ---------------------------------------------------------------------------
__global__ __launch_bounds__(256)
void chunkrec_kernel(float* __restrict__ states, const float* __restrict__ total) {
  const int id = blockIdx.x * blockDim.x + threadIdx.x;
  const int pn = id & (HEADDIM * DSTATE - 1);
  const int bh = id >> 13;
  if (bh >= BATCH * NHEADS) return;
  const int b = bh >> 4, h = bh & (NHEADS - 1);
  float S = 0.f;
  for (int z = 0; z < NCHUNK; ++z) {
    const int bch = (b * NCHUNK + z) * NHEADS + h;
    const size_t idx = (size_t)bch * HEADDIM * DSTATE + pn;
    const float tmp = states[idx];
    states[idx] = S;
    S = expf(total[bch]) * S + tmp;
  }
}

// ---------------------------------------------------------------------------
// fused Y = (Y_diag + Y_off + D*x) * silu(z), per (b,c,h,qtile=64).
// ---------------------------------------------------------------------------
__global__ __launch_bounds__(256)
void ychunk_kernel(const float* __restrict__ proj, const float* __restrict__ dt,
                   const float* __restrict__ acs, const float* __restrict__ prev,
                   const float* __restrict__ D_skip, float* __restrict__ ybuf) {
  __shared__ __align__(16) _Float16 Cs [64 * 72];    // [q][n]
  __shared__ __align__(16) _Float16 Cs2[64 * 72];    // [q][n]  C * exp(Acs_q)
  __shared__ __align__(16) _Float16 Pt [128 * 72];   // [p][n]  prev (row-major)
  __shared__ __align__(16) _Float16 Bkt[32 * 72];    // [k][n]  B k-slice
  __shared__ __align__(16) _Float16 Gs [64 * 40];    // [q][k]
  __shared__ __align__(16) _Float16 xs [128 * 40];   // [p][k]
  __shared__ float AcsQ[64];
  __shared__ float AcsK[CHUNK];
  __shared__ float dtK [CHUNK];

  const int blk = blockIdx.x;
  const int qt = blk & 3;
  const int h  = (blk >> 2) & (NHEADS - 1);
  const int c  = (blk >> 6) & (NCHUNK - 1);
  const int b  = blk >> 10;
  const int t = threadIdx.x, lane = t & 31, wave = t >> 5;
  const int qbase = qt * 64;
  const int lbase = c * CHUNK;
  const int bch = (b * NCHUNK + c) * NHEADS + h;

  // ---- stage scalars ----
  AcsK[t] = acs[(size_t)bch * CHUNK + t];
  dtK[t]  = dt[((size_t)(b * LSEQ + lbase + t)) * NHEADS + h];
  if (t < 64) AcsQ[t] = acs[(size_t)bch * CHUNK + qbase + t];
  __syncthreads();

  // ---- stage C / C*exp(Acs_q) as [q][n], prev as [p][n]; all b64 stores ----
  {
    const int q = t >> 2, n0 = (t & 3) * 16;
    const size_t row = (size_t)(b * LSEQ + lbase + qbase + q) * DINPROJ + 2 * DINNER + DSTATE;
    const float eq = expf(AcsQ[q]);
#pragma unroll
    for (int i = 0; i < 4; ++i) {
      const float4 v = *(const float4*)(proj + row + n0 + 4 * i);
      *(v4h*)(Cs  + q * 72 + n0 + 4 * i) = pack4(v);
      *(v4h*)(Cs2 + q * 72 + n0 + 4 * i) = pack4s(v, eq);
    }
  }
  {
    const int p = t >> 1, n0 = (t & 1) * 32;
    const size_t sbase = (size_t)bch * HEADDIM * DSTATE + (size_t)p * DSTATE;
#pragma unroll
    for (int i = 0; i < 8; ++i) {
      const float4 v = *(const float4*)(prev + sbase + n0 + 4 * i);
      *(v4h*)(Pt + p * 72 + n0 + 4 * i) = pack4(v);
    }
  }
  __syncthreads();

  // wave -> 16-row q strip (qw) x 64-col p half (pw)
  const int qw = wave >> 1, pw = wave & 1;
  v8f acc[4] = {v8f{}, v8f{}, v8f{}, v8f{}};

  // ---- Y_off: (Cs2) x prev^T, K-dim = n (2 steps of 32) ----
#pragma unroll
  for (int ns = 0; ns < 2; ++ns) {
    const v16h a = load_a_frag(Cs2 + ns * 32, 72, qw * 16 + (lane & 15), lane);
#pragma unroll
    for (int j = 0; j < 4; ++j) {
      const v16h bf = load_b_frag(Pt + ns * 32, 72,
                                  pw * 64 + j * 16 + (lane & 15), lane);
      acc[j] = wmma_f16(a, bf, acc[j]);
    }
  }

  // ---- Y_diag over causal k tiles ----
  const int ktmax = (qbase + 63) / 32;
  for (int kt = 0; kt <= ktmax; ++kt) {
    const int kbase = kt * 32;
    // stage B [k][n] (b64 stores) and x [p][k] (b16 scatter)
    {
      const int kk = t >> 3;
      const size_t row = (size_t)(b * LSEQ + lbase + kbase + kk) * DINPROJ;
      const int n0 = (t & 7) * 8;
#pragma unroll
      for (int i = 0; i < 2; ++i) {
        const float4 v = *(const float4*)(proj + row + 2 * DINNER + n0 + 4 * i);
        *(v4h*)(Bkt + kk * 72 + n0 + 4 * i) = pack4(v);
      }
      const int p0 = (t & 7) * 16;
#pragma unroll
      for (int i = 0; i < 4; ++i) {
        const float4 v = *(const float4*)(proj + row + DINNER + h * HEADDIM + p0 + 4 * i);
        xs[(p0 + 4 * i + 0) * 40 + kk] = (_Float16)v.x;
        xs[(p0 + 4 * i + 1) * 40 + kk] = (_Float16)v.y;
        xs[(p0 + 4 * i + 2) * 40 + kk] = (_Float16)v.z;
        xs[(p0 + 4 * i + 3) * 40 + kk] = (_Float16)v.w;
      }
    }
    __syncthreads();

    // G tile: each wave one 16x16 of the 64x32 G block (qg x kg)
    {
      const int qg = wave >> 1, kg = wave & 1;
      v8f g = {};
#pragma unroll
      for (int ns = 0; ns < 2; ++ns) {
        const v16h a  = load_a_frag(Cs + ns * 32, 72, qg * 16 + (lane & 15), lane);
        const v16h bf = load_b_frag(Bkt + ns * 32, 72, kg * 16 + (lane & 15), lane);
        g = wmma_f16(a, bf, g);
      }
      // scale by dt_k * exp(Acs_q - Acs_k) * causal mask, write [q][k]
      const int kl = kg * 16 + (lane & 15);
      const int kabs = kbase + kl;
      const int qb8 = qg * 16 + 8 * (lane >> 4);
#pragma unroll
      for (int r = 0; r < 8; ++r) {
        const int ql = qb8 + r;
        const int qabs = qbase + ql;
        float val = 0.f;
        if (kabs <= qabs)
          val = g[r] * dtK[kabs] * expf(AcsQ[ql] - AcsK[kabs]);
        Gs[ql * 40 + kl] = (_Float16)val;
      }
    }
    __syncthreads();

    // Y += G x x
    {
      const v16h a = load_a_frag(Gs, 40, qw * 16 + (lane & 15), lane);
#pragma unroll
      for (int j = 0; j < 4; ++j) {
        const v16h bf = load_b_frag(xs, 40, pw * 64 + j * 16 + (lane & 15), lane);
        acc[j] = wmma_f16(a, bf, acc[j]);
      }
    }
    __syncthreads();
  }

  // ---- epilogue: +D*x, gate with silu(z), write ybuf ----
  const float Dh = D_skip[h];
#pragma unroll
  for (int j = 0; j < 4; ++j) {
    const int p = pw * 64 + j * 16 + (lane & 15);
    const int qb8 = qw * 16 + 8 * (lane >> 4);
#pragma unroll
    for (int r = 0; r < 8; ++r) {
      const int ql = qb8 + r;
      const int l = lbase + qbase + ql;
      const size_t row = (size_t)(b * LSEQ + l) * DINPROJ;
      const float xv = proj[row + DINNER + h * HEADDIM + p];
      const float zv = proj[row + h * HEADDIM + p];
      float yv = acc[j][r] + Dh * xv;
      yv *= zv / (1.f + expf(-zv));
      ybuf[(size_t)(b * LSEQ + l) * DINNER + h * HEADDIM + p] = yv;
    }
  }
}

// ---------------------------------------------------------------------------
// launch
// ---------------------------------------------------------------------------
extern "C" void kernel_launch(void* const* d_in, const int* in_sizes, int n_in,
                              void* d_out, int out_size, void* d_ws, size_t ws_size,
                              hipStream_t stream) {
  const float* x       = (const float*)d_in[0];
  const float* W_in    = (const float*)d_in[1];
  const float* conv_w  = (const float*)d_in[2];
  const float* conv_b  = (const float*)d_in[3];
  const float* A_log   = (const float*)d_in[4];
  const float* dt_bias = (const float*)d_in[5];
  const float* D_skip  = (const float*)d_in[6];
  const float* W_out   = (const float*)d_in[7];
  float* out = (float*)d_out;

  float* ws     = (float*)d_ws;
  float* proj   = ws + OFF_PROJ;
  float* dtb    = ws + OFF_DT;
  float* acs    = ws + OFF_ACS;
  float* total  = ws + OFF_TOT;
  float* states = ws + OFF_ST;
  float* ybuf   = ws + OFF_Y;

  gemm_tn<<<dim3((DINPROJ + 63) / 64, MTOT / 128), 256, 0, stream>>>(
      x, W_in, proj, MTOT, DINPROJ, DMODEL);

  conv_silu_kernel<<<(BATCH * CONVDIM + 255) / 256, 256, 0, stream>>>(
      proj, conv_w, conv_b);

  dt_kernel<<<(BATCH * LSEQ * NHEADS) / 256, 256, 0, stream>>>(proj, dt_bias, dtb);

  cumsum_kernel<<<(BATCH * NCHUNK * NHEADS + 255) / 256, 256, 0, stream>>>(
      dtb, A_log, acs, total);

  states_kernel<<<BATCH * NCHUNK * NHEADS, 256, 0, stream>>>(
      proj, dtb, acs, total, states);

  chunkrec_kernel<<<(BATCH * NHEADS * HEADDIM * DSTATE) / 256, 256, 0, stream>>>(
      states, total);

  ychunk_kernel<<<BATCH * NCHUNK * NHEADS * 4, 256, 0, stream>>>(
      proj, dtb, acs, states, D_skip, ybuf);

  gemm_tn<<<dim3(DMODEL / 64, MTOT / 128), 256, 0, stream>>>(
      ybuf, W_out, out, MTOT, DMODEL, DINNER);
}
